// MultiheadSelfAttention_86870008529042
// MI455X (gfx1250) — matrix operand order repaired and checked by reading, not compiled, not run on hardware
//
#include <hip/hip_runtime.h>
#include <hip/hip_bf16.h>

// ---------------------------------------------------------------------------
// Fused BatchNorm(F) + QKV projection + multi-head self-attention for
// B=8, F=1024, E=1024, H=16, hd=64 on gfx1250 (wave32, WMMA bf16).
// ---------------------------------------------------------------------------

typedef __attribute__((ext_vector_type(16))) __bf16 v16bf;
typedef __attribute__((ext_vector_type(8)))  float  v8f;
typedef __attribute__((ext_vector_type(4)))  unsigned int u32x4;
typedef __attribute__((ext_vector_type(2)))  unsigned int u32x2;

#define BQ   8
#define FQ   1024
#define EQ   1024
#define HQ   16
#define HD   64
#define O3   3072                 // 3*E
#define ROWS 8192                 // B*F

__device__ __forceinline__ unsigned short f2bf_bits(float f) {
    unsigned int u = __builtin_bit_cast(unsigned int, f);
    u += 0x7fffu + ((u >> 16) & 1u);          // round-to-nearest-even
    return (unsigned short)(u >> 16);
}
__device__ __forceinline__ __bf16 f2bf(float f) {
    unsigned short s = f2bf_bits(f);
    return __builtin_bit_cast(__bf16, s);
}

union FragU { u32x4 q[2]; v16bf v; };

// B-fragment (32x16 bf16): 16 contiguous halves per lane.
__device__ __forceinline__ v16bf load_b_frag(const __bf16* p) {
    FragU u;
    u.q[0] = *(const u32x4*)(p);
    u.q[1] = *(const u32x4*)(p + 8);
    return u.v;
}
// A-fragment (16x32 bf16): 16 contiguous halves per lane.
__device__ __forceinline__ v16bf load_a_frag(const __bf16* p) {
    FragU u;
    u.q[0] = *(const u32x4*)(p);
    u.q[1] = *(const u32x4*)(p + 8);
    return u.v;
}

__device__ __forceinline__ v8f wmma_bf16(v16bf a, v16bf b, v8f c) {
    // (neg_a, A, neg_b, B, c_mod, C, reuse_a, reuse_b)
    return __builtin_amdgcn_wmma_f32_16x16x32_bf16(false, a, false, b,
                                                   (short)0, c, false, false);
}

// ---------------------------------------------------------------------------
// 1) BatchNorm statistics: mean/rstd per feature f, reduced over (B, E).
// ---------------------------------------------------------------------------
__global__ __launch_bounds__(256) void bn_stats_kernel(const float* __restrict__ x,
                                                       float* __restrict__ mean,
                                                       float* __restrict__ rstd) {
    const int f = blockIdx.x;
    float s = 0.f, s2 = 0.f;
    for (int i = threadIdx.x; i < BQ * EQ; i += 256) {
        int bb = i >> 10, e = i & 1023;
        float v = x[((size_t)bb * FQ + f) * EQ + e];
        s += v; s2 += v * v;
    }
    __shared__ float r1[256], r2[256];
    r1[threadIdx.x] = s; r2[threadIdx.x] = s2;
    __syncthreads();
    for (int st = 128; st > 0; st >>= 1) {
        if ((int)threadIdx.x < st) {
            r1[threadIdx.x] += r1[threadIdx.x + st];
            r2[threadIdx.x] += r2[threadIdx.x + st];
        }
        __syncthreads();
    }
    if (threadIdx.x == 0) {
        float m   = r1[0] * (1.f / (BQ * EQ));
        float var = r2[0] * (1.f / (BQ * EQ)) - m * m;
        mean[f] = m;
        rstd[f] = rsqrtf(var + 1e-5f);
    }
}

// ---------------------------------------------------------------------------
// 2) Normalize + affine, convert to bf16  Xn[B*F, E]
// ---------------------------------------------------------------------------
__global__ __launch_bounds__(256) void bn_apply_kernel(const float* __restrict__ x,
                                                       const float* __restrict__ mean,
                                                       const float* __restrict__ rstd,
                                                       const float* __restrict__ gamma,
                                                       const float* __restrict__ beta,
                                                       __bf16* __restrict__ Xn) {
    size_t idx = ((size_t)blockIdx.x * 256 + threadIdx.x) * 4;
    int row = (int)(idx >> 10);         // b*F + f   (E = 1024)
    int f   = row & (FQ - 1);
    float m = mean[f], r = rstd[f], g = gamma[f], bt = beta[f];
    const float4 v = *(const float4*)(x + idx);
    float a0 = (v.x - m) * r * g + bt;
    float a1 = (v.y - m) * r * g + bt;
    float a2 = (v.z - m) * r * g + bt;
    float a3 = (v.w - m) * r * g + bt;
    u32x2 o;
    o.x = (unsigned)f2bf_bits(a0) | ((unsigned)f2bf_bits(a1) << 16);
    o.y = (unsigned)f2bf_bits(a2) | ((unsigned)f2bf_bits(a3) << 16);
    *(u32x2*)(Xn + idx) = o;
}

// ---------------------------------------------------------------------------
// 3) Weight fp32 -> bf16
// ---------------------------------------------------------------------------
__global__ __launch_bounds__(256) void wcvt_kernel(const float* __restrict__ W,
                                                   __bf16* __restrict__ Wb) {
    size_t idx = ((size_t)blockIdx.x * 256 + threadIdx.x) * 4;
    const float4 v = *(const float4*)(W + idx);
    u32x2 o;
    o.x = (unsigned)f2bf_bits(v.x) | ((unsigned)f2bf_bits(v.y) << 16);
    o.y = (unsigned)f2bf_bits(v.z) | ((unsigned)f2bf_bits(v.w) << 16);
    *(u32x2*)(Wb + idx) = o;
}

// ---------------------------------------------------------------------------
// 4) QKV GEMM: qkv[r, o] = sum_k Xn[r, k] * W[o, k] + bias[o]
//    Per-wave 64x64 tile, K-loop step 32, V_WMMA_F32_16X16X32_BF16.
//    Epilogue: bf16 qkv; V columns (c%192 >= 128) also scattered transposed
//    into Vt[bh][d][f] so attention B-fragments are contiguous.
// ---------------------------------------------------------------------------
__global__ __launch_bounds__(128) void qkv_gemm_kernel(const __bf16* __restrict__ Xn,
                                                       const __bf16* __restrict__ Wb,
                                                       const float* __restrict__ bias,
                                                       __bf16* __restrict__ qkvb,
                                                       __bf16* __restrict__ Vt) {
    const int lane = threadIdx.x & 31;
    const int wave = threadIdx.x >> 5;          // 0..3
    const int ln   = lane & 15;
    const int hi   = lane >> 4;                 // 0/1 half-wave
    const int m0 = blockIdx.x * 128 + (wave & 1) * 64;
    const int n0 = blockIdx.y * 128 + (wave >> 1) * 64;

    v8f acc[4][4];
#pragma unroll
    for (int i = 0; i < 4; ++i)
#pragma unroll
        for (int j = 0; j < 4; ++j) acc[i][j] = {};

    for (int k0 = 0; k0 < EQ; k0 += 32) {
        v16bf a[4], b[4];
#pragma unroll
        for (int i = 0; i < 4; ++i)
            a[i] = load_a_frag(Xn + (size_t)(m0 + 16 * i + ln) * EQ + k0 + hi * 16);
#pragma unroll
        for (int j = 0; j < 4; ++j)
            b[j] = load_b_frag(Wb + (size_t)(n0 + 16 * j + ln) * EQ + k0 + hi * 16);
#pragma unroll
        for (int i = 0; i < 4; ++i)
#pragma unroll
            for (int j = 0; j < 4; ++j)
                acc[i][j] = wmma_bf16(a[i], b[j], acc[i][j]);
    }

#pragma unroll
    for (int j = 0; j < 4; ++j) {
        const int col = n0 + 16 * j + ln;
        const float bv = bias[col];
        const int hh = col / 192;               // head
        const int cc = col - hh * 192;          // 0..191 within head
#pragma unroll
        for (int i = 0; i < 4; ++i) {
#pragma unroll
            for (int r = 0; r < 8; ++r) {
                const int row = m0 + 16 * i + r + hi * 8;   // b*F + f
                __bf16 hval = f2bf(acc[i][j][r] + bv);
                qkvb[(size_t)row * O3 + col] = hval;
                if (cc >= 128) {                // V element -> transposed copy
                    const int d  = cc - 128;
                    const int bb = row >> 10;
                    const int f  = row & (FQ - 1);
                    Vt[(((size_t)(bb * HQ + hh) * HD + d) * FQ) + f] = hval;
                }
            }
        }
    }
}

// ---------------------------------------------------------------------------
// 5) Flash attention: per wave, 16 query rows of one (b,h).
//    S = Q K^T * 0.125 via WMMA, fp32 online softmax (row state in the C/D
//    layout: lane holds 8 rows M = r + 8*hi, column N = lane&15; row
//    reductions via width-16 shfl_xor), P -> bf16 through LDS, O += P V.
// ---------------------------------------------------------------------------
__global__ __launch_bounds__(128) void attn_kernel(const __bf16* __restrict__ qkvb,
                                                   const __bf16* __restrict__ Vt,
                                                   float* __restrict__ out) {
    const int lane = threadIdx.x & 31;
    const int wave = threadIdx.x >> 5;
    const int ln   = lane & 15;
    const int hi   = lane >> 4;
    const int bh = blockIdx.x;
    const int b  = bh >> 4;
    const int h  = bh & 15;
    const int q0 = blockIdx.y * 64 + wave * 16;

    __shared__ __bf16 plds_all[4][16][64];
    __bf16 (*plds)[64] = plds_all[wave];

    // Q A-fragments (rows q0..q0+15, d = 0..63) — head block is interleaved:
    // head h owns qkv columns [h*192, h*192+192), q at +0, k at +64, v at +128.
    const __bf16* qbase = qkvb + (size_t)(b * FQ + q0 + ln) * O3 + h * 192;
    v16bf aQ0 = load_a_frag(qbase + hi * 16);
    v16bf aQ1 = load_a_frag(qbase + 32 + hi * 16);

    const __bf16* kbase = qkvb + (size_t)(b * FQ) * O3 + h * 192 + 64;
    const __bf16* vtb   = Vt + (size_t)bh * HD * FQ;

    float mrow[8], lrow[8];
    v8f o[4];
#pragma unroll
    for (int r = 0; r < 8; ++r) { mrow[r] = -3.0e38f; lrow[r] = 0.f; }
#pragma unroll
    for (int t = 0; t < 4; ++t) o[t] = {};

    for (int j = 0; j < FQ; j += 64) {
        // ---- S = Q K^T for 64 keys (4 N-tiles) ----
        v8f s[4];
#pragma unroll
        for (int t = 0; t < 4; ++t) {
            const __bf16* kp = kbase + (size_t)(j + t * 16 + ln) * O3;
            v16bf bk0 = load_b_frag(kp + hi * 16);
            v16bf bk1 = load_b_frag(kp + 32 + hi * 16);
            v8f z = {};
            z = wmma_bf16(aQ0, bk0, z);
            z = wmma_bf16(aQ1, bk1, z);
            s[t] = z;
        }
        // ---- scale + online softmax ----
#pragma unroll
        for (int r = 0; r < 8; ++r) {
            float s0 = s[0][r] * 0.125f, s1 = s[1][r] * 0.125f;
            float s2 = s[2][r] * 0.125f, s3 = s[3][r] * 0.125f;
            float rm = fmaxf(fmaxf(s0, s1), fmaxf(s2, s3));
            rm = fmaxf(rm, __shfl_xor(rm, 1, 16));
            rm = fmaxf(rm, __shfl_xor(rm, 2, 16));
            rm = fmaxf(rm, __shfl_xor(rm, 4, 16));
            rm = fmaxf(rm, __shfl_xor(rm, 8, 16));
            float mnew  = fmaxf(mrow[r], rm);
            float alpha = __expf(mrow[r] - mnew);
            mrow[r] = mnew;
            float p0 = __expf(s0 - mnew), p1 = __expf(s1 - mnew);
            float p2 = __expf(s2 - mnew), p3 = __expf(s3 - mnew);
            s[0][r] = p0; s[1][r] = p1; s[2][r] = p2; s[3][r] = p3;
            float rs = p0 + p1 + p2 + p3;
            rs += __shfl_xor(rs, 1, 16);
            rs += __shfl_xor(rs, 2, 16);
            rs += __shfl_xor(rs, 4, 16);
            rs += __shfl_xor(rs, 8, 16);
            lrow[r] = lrow[r] * alpha + rs;
#pragma unroll
            for (int t = 0; t < 4; ++t) o[t][r] *= alpha;
        }
        // ---- P (D-layout) -> bf16 in LDS (row-major 16x64) ----
#pragma unroll
        for (int t = 0; t < 4; ++t)
#pragma unroll
            for (int r = 0; r < 8; ++r)
                plds[r + hi * 8][t * 16 + ln] = f2bf(s[t][r]);
        __asm__ volatile("s_wait_dscnt 0x0" ::: "memory");

        // ---- re-read P as A-fragments (keys 0..31, 32..63 of this tile) ----
        v16bf aP0 = load_a_frag(&plds[ln][hi * 16]);
        v16bf aP1 = load_a_frag(&plds[ln][32 + hi * 16]);

        // ---- O += P V  (contraction over 64 keys, N = hd) ----
#pragma unroll
        for (int t = 0; t < 4; ++t) {
            const __bf16* vp = vtb + (size_t)(t * 16 + ln) * FQ + j;
            v16bf bv0 = load_b_frag(vp + hi * 16);
            v16bf bv1 = load_b_frag(vp + 32 + hi * 16);
            o[t] = wmma_bf16(aP0, bv0, o[t]);
            o[t] = wmma_bf16(aP1, bv1, o[t]);
        }
    }

    // ---- epilogue: out[b, f, h*64 + d] = O / l ----
#pragma unroll
    for (int t = 0; t < 4; ++t)
#pragma unroll
        for (int r = 0; r < 8; ++r) {
            const int m = r + hi * 8;
            out[((size_t)(b * FQ + q0 + m)) * EQ + h * HD + t * 16 + ln] =
                o[t][r] / lrow[r];
        }
}

// ---------------------------------------------------------------------------
extern "C" void kernel_launch(void* const* d_in, const int* in_sizes, int n_in,
                              void* d_out, int out_size, void* d_ws, size_t ws_size,
                              hipStream_t stream) {
    const float* x     = (const float*)d_in[0];
    const float* gamma = (const float*)d_in[1];
    const float* beta  = (const float*)d_in[2];
    const float* W     = (const float*)d_in[3];
    const float* bqkv  = (const float*)d_in[4];
    float* out = (float*)d_out;

    char* ws = (char*)d_ws;
    float* mean = (float*)ws;              ws += (size_t)FQ * 4;
    float* rstd = (float*)ws;              ws += (size_t)FQ * 4;
    __bf16* Xn   = (__bf16*)ws;            ws += (size_t)ROWS * EQ * 2;    // 16 MB
    __bf16* Wb   = (__bf16*)ws;            ws += (size_t)O3 * EQ * 2;      //  6 MB
    __bf16* qkvb = (__bf16*)ws;            ws += (size_t)ROWS * O3 * 2;    // 48 MB
    __bf16* Vt   = (__bf16*)ws;            // B*H * hd * F bf16           // 16 MB

    bn_stats_kernel<<<FQ, 256, 0, stream>>>(x, mean, rstd);
    bn_apply_kernel<<<(ROWS * EQ) / (256 * 4), 256, 0, stream>>>(x, mean, rstd,
                                                                 gamma, beta, Xn);
    wcvt_kernel<<<(O3 * EQ) / (256 * 4), 256, 0, stream>>>(W, Wb);
    qkv_gemm_kernel<<<dim3(ROWS / 128, O3 / 128), 128, 0, stream>>>(Xn, Wb, bqkv,
                                                                    qkvb, Vt);
    attn_kernel<<<dim3(BQ * HQ, FQ / 64), 128, 0, stream>>>(qkvb, Vt, out);
}
